// TtCodeGenAttention_14834817040847
// MI455X (gfx1250) — matrix operand order repaired and checked by reading, not compiled, not run on hardware
//
#include <hip/hip_runtime.h>

typedef _Float16 half_t;
typedef __attribute__((ext_vector_type(16))) _Float16 v16h;
typedef __attribute__((ext_vector_type(8)))  _Float16 v8h;
typedef __attribute__((ext_vector_type(8)))  float    v8f;
typedef __attribute__((ext_vector_type(4)))  float    v4f;

union V16 { v16h v; v8h h[2]; };

#define WMMA_F16(A, B, C) \
    __builtin_amdgcn_wmma_f32_16x16x32_f16(false, (A), false, (B), (short)0, (C), false, false)

// ---------------------------------------------------------------------------
// f32 -> f16 conversion, 8 elements/thread (pure b128 traffic)
// ---------------------------------------------------------------------------
__global__ void f32_to_f16_kernel(const float* __restrict__ src,
                                  half_t* __restrict__ dst, int n8) {
    int i = blockIdx.x * blockDim.x + threadIdx.x;   // one per 8 elements
    if (i >= n8) return;
    v4f a = *(const v4f*)(src + (size_t)i * 8);
    v4f b = *(const v4f*)(src + (size_t)i * 8 + 4);
    v8h o;
#pragma unroll
    for (int j = 0; j < 4; ++j) { o[j] = (half_t)a[j]; o[4 + j] = (half_t)b[j]; }
    *(v8h*)(dst + (size_t)i * 8) = o;
}

// ---------------------------------------------------------------------------
// GEMM: C[M,N] = A[M,K] * W[N,K]^T   (A row-major, W row-major N x K)
// Block: 256 threads = 8 waves; each wave computes a 32x64 tile
// (2 A-fragments reuse each B-fragment twice).  Grid: (N/64, M/256).
// ---------------------------------------------------------------------------
template <typename OutT>
__global__ void gemm_wt_kernel(const half_t* __restrict__ A,
                               const half_t* __restrict__ W,
                               OutT* __restrict__ C,
                               int M, int N, int K) {
    const int lane = threadIdx.x & 31;
    const int wave = threadIdx.x >> 5;
    const int r  = lane & 15;      // row/col within 16-tile
    const int hf = lane >> 4;      // half-wave selector

    const int row0 = (blockIdx.y * 8 + wave) * 32;
    const int col0 = blockIdx.x * 64;
    if (row0 >= M) return;

    v8f acc[2][4] = {};
    const half_t* Arow0 = A + (size_t)(row0 + r) * K;
    const half_t* Arow1 = Arow0 + (size_t)16 * K;

    for (int k0 = 0; k0 < K; k0 += 32) {
        V16 a0, a1;
        a0.h[0] = *(const v8h*)(Arow0 + k0 + hf * 8);
        a0.h[1] = *(const v8h*)(Arow0 + k0 + 16 + hf * 8);
        a1.h[0] = *(const v8h*)(Arow1 + k0 + hf * 8);
        a1.h[1] = *(const v8h*)(Arow1 + k0 + 16 + hf * 8);
#pragma unroll
        for (int t = 0; t < 4; ++t) {
            const half_t* Wrow = W + (size_t)(col0 + t * 16 + r) * K + k0 + hf * 8;
            V16 b;
            b.h[0] = *(const v8h*)(Wrow);
            b.h[1] = *(const v8h*)(Wrow + 16);
            acc[0][t] = WMMA_F16(a0.v, b.v, acc[0][t]);
            acc[1][t] = WMMA_F16(a1.v, b.v, acc[1][t]);
        }
    }

    // C/D layout: lane L, vgpr j -> row = j + 8*(L/16), col = L%16
#pragma unroll
    for (int m = 0; m < 2; ++m)
#pragma unroll
        for (int t = 0; t < 4; ++t)
#pragma unroll
            for (int j = 0; j < 8; ++j) {
                int row = row0 + m * 16 + j + 8 * hf;
                int col = col0 + t * 16 + r;
                C[(size_t)row * N + col] = (OutT)acc[m][t][j];
            }
}

// ---------------------------------------------------------------------------
// RoPE + head split. qkv is (B,S,3072) f16; per MP partition of 768:
//   [0,256)=Q  [256,512)=V  [512,768)=K ; head h: mp=h>>2, sub=h&3.
// Writes Qh,Kh as [b,h,s,d] f16 (Q pre-scaled by 1/8), V transposed [b,h,d,s].
// ---------------------------------------------------------------------------
__global__ void rope_split_kernel(const half_t* __restrict__ qkv,
                                  half_t* __restrict__ Qh,
                                  half_t* __restrict__ Kh,
                                  half_t* __restrict__ Vt) {
    int idx = blockIdx.x * blockDim.x + threadIdx.x;   // B*H*S*D = 4194304
    int d = idx & 63;
    int s = (idx >> 6) & 2047;
    int h = (idx >> 17) & 15;
    int b = idx >> 21;
    int mp = h >> 2, sub = h & 3;

    size_t base = ((size_t)(b * 2048 + s)) * 3072 + mp * 768 + sub * 64;
    float q = (float)qkv[base + d];
    float v = (float)qkv[base + 256 + d];
    float k = (float)qkv[base + 512 + d];

    float qr = q, kr = k;
    if (d < 32) {
        float inv = __powf(10000.0f, -(float)(d & ~1) * (1.0f / 32.0f));
        float ang = (float)s * inv;
        float sv = __sinf(ang), cv = __cosf(ang);
        if (d & 1) {
            float qp = (float)qkv[base + d - 1];
            float kp = (float)qkv[base + 512 + d - 1];
            qr = q * cv + qp * sv;
            kr = k * cv + kp * sv;
        } else {
            float qn = (float)qkv[base + d + 1];
            float kn = (float)qkv[base + 512 + d + 1];
            qr = q * cv - qn * sv;
            kr = k * cv - kn * sv;
        }
    }
    size_t bh = (size_t)(b * 16 + h);
    Qh[(bh * 2048 + s) * 64 + d] = (half_t)(qr * 0.125f);  // fold 1/SCALE into Q
    Kh[(bh * 2048 + s) * 64 + d] = (half_t)kr;
    Vt[(bh * 64 + d) * 2048 + s] = (half_t)v;
}

// ---------------------------------------------------------------------------
// One 32-key flash-attention chunk.  MASKED is a compile-time flag so the
// bulk (fully-causal-valid) loop carries zero mask VALU ops; only the single
// diagonal chunk per q-tile pays for v_cmp/v_cndmask.
// ---------------------------------------------------------------------------
template <bool MASKED>
__device__ __forceinline__ void attn_chunk(
        int kc, int q0, int r, int hf, int bh,
        const half_t* __restrict__ Kh, const half_t* __restrict__ Vt,
        const V16& qa0, const V16& qa1,
        half_t* __restrict__ P,
        float m_i[8], float l_i[8], v8f o[4]) {
    v8f sc[2];
#pragma unroll
    for (int t = 0; t < 2; ++t) {
        const int n0 = kc + t * 16;
        const half_t* Krow = Kh + ((size_t)bh * 2048 + n0 + r) * 64;
        V16 kb0, kb1;
        kb0.h[0] = *(const v8h*)(Krow + hf * 8);
        kb0.h[1] = *(const v8h*)(Krow + 16 + hf * 8);
        kb1.h[0] = *(const v8h*)(Krow + 32 + hf * 8);
        kb1.h[1] = *(const v8h*)(Krow + 48 + hf * 8);
        v8f s = {};
        s = WMMA_F16(qa0.v, kb0.v, s);
        s = WMMA_F16(qa1.v, kb1.v, s);
        if (MASKED) {
#pragma unroll
            for (int j = 0; j < 8; ++j) {
                int query = q0 + j + 8 * hf;
                int key   = n0 + r;
                if (key > query) s[j] = -1.0e7f;   // MASK_VALUE
            }
        }
        sc[t] = s;
    }

    float rmax[8];
#pragma unroll
    for (int j = 0; j < 8; ++j) rmax[j] = fmaxf(sc[0][j], sc[1][j]);
#pragma unroll
    for (int j = 0; j < 8; ++j)
        for (int mask = 1; mask < 16; mask <<= 1)
            rmax[j] = fmaxf(rmax[j], __shfl_xor(rmax[j], mask));

    float m_new[8], rsum[8];
#pragma unroll
    for (int j = 0; j < 8; ++j) { m_new[j] = fmaxf(m_i[j], rmax[j]); rsum[j] = 0.0f; }

#pragma unroll
    for (int t = 0; t < 2; ++t)
#pragma unroll
        for (int j = 0; j < 8; ++j) {
            float p = __expf(sc[t][j] - m_new[j]);
            rsum[j] += p;
            P[(j + 8 * hf) * 32 + t * 16 + r] = (half_t)p;   // C-layout -> LDS
        }

#pragma unroll
    for (int j = 0; j < 8; ++j)
        for (int mask = 1; mask < 16; mask <<= 1)
            rsum[j] += __shfl_xor(rsum[j], mask);

#pragma unroll
    for (int j = 0; j < 8; ++j) {
        float corr = __expf(m_i[j] - m_new[j]);
        l_i[j] = l_i[j] * corr + rsum[j];
        m_i[j] = m_new[j];
#pragma unroll
        for (int td = 0; td < 4; ++td) o[td][j] *= corr;
    }

    // reload P as an A-fragment (16 x 32 keys); same-wave DS ops are in-order
    V16 pa;
    pa.h[0] = *(const v8h*)(P + r * 32 + hf * 8);
    pa.h[1] = *(const v8h*)(P + r * 32 + 16 + hf * 8);

#pragma unroll
    for (int td = 0; td < 4; ++td) {
        const half_t* Vcol = Vt + ((size_t)bh * 64 + td * 16 + r) * 2048 + kc + hf * 8;
        V16 vb;
        vb.h[0] = *(const v8h*)(Vcol);
        vb.h[1] = *(const v8h*)(Vcol + 16);
        o[td] = WMMA_F16(pa.v, vb.v, o[td]);
    }
}

// ---------------------------------------------------------------------------
// Causal flash attention. Block = 128 threads = 4 independent waves; each
// wave owns one 16-query tile. Bulk chunks (keys <= q0) run mask-free; the
// single diagonal chunk km = (q0/32)*32 applies the causal mask.
// Output written [b,s, h*64+d] f16 for the final projection.
// ---------------------------------------------------------------------------
__global__ void attn_kernel(const half_t* __restrict__ Qh,
                            const half_t* __restrict__ Kh,
                            const half_t* __restrict__ Vt,
                            half_t* __restrict__ Oout) {
    __shared__ half_t Pbuf[4][16 * 32];

    const int lane = threadIdx.x & 31;
    const int wave = threadIdx.x >> 5;
    const int r  = lane & 15;
    const int hf = lane >> 4;

    const int gw = blockIdx.x * 4 + wave;      // 0 .. 4095 q-tiles
    const int qt = gw & 127;                   // S/16 = 128 per (b,h)
    const int bh = gw >> 7;
    const int q0 = qt * 16;
    half_t* P = &Pbuf[wave][0];

    // Q A-fragments for d in [0,32) and [32,64); resident for the whole sweep
    const half_t* Qrow = Qh + ((size_t)bh * 2048 + q0 + r) * 64;
    V16 qa0, qa1;
    qa0.h[0] = *(const v8h*)(Qrow + hf * 8);
    qa0.h[1] = *(const v8h*)(Qrow + 16 + hf * 8);
    qa1.h[0] = *(const v8h*)(Qrow + 32 + hf * 8);
    qa1.h[1] = *(const v8h*)(Qrow + 48 + hf * 8);

    float m_i[8], l_i[8];
#pragma unroll
    for (int j = 0; j < 8; ++j) { m_i[j] = -3.0e38f; l_i[j] = 0.0f; }
    v8f o[4] = {};

    const int km = (q0 >> 5) << 5;             // diagonal chunk start
    for (int kc = 0; kc < km; kc += 32)        // fully unmasked bulk
        attn_chunk<false>(kc, q0, r, hf, bh, Kh, Vt, qa0, qa1, P, m_i, l_i, o);
    attn_chunk<true>(km, q0, r, hf, bh, Kh, Vt, qa0, qa1, P, m_i, l_i, o);

    const int b = bh >> 4, h = bh & 15;
#pragma unroll
    for (int td = 0; td < 4; ++td)
#pragma unroll
        for (int j = 0; j < 8; ++j) {
            int srow = q0 + j + 8 * hf;
            int feat = h * 64 + td * 16 + r;
            Oout[((size_t)(b * 2048 + srow)) * 1024 + feat] = (half_t)(o[td][j] / l_i[j]);
        }
}

// ---------------------------------------------------------------------------
// Host-side launch
// ---------------------------------------------------------------------------
extern "C" void kernel_launch(void* const* d_in, const int* in_sizes, int n_in,
                              void* d_out, int out_size, void* d_ws, size_t ws_size,
                              hipStream_t stream) {
    (void)in_sizes; (void)n_in; (void)out_size; (void)ws_size;
    const int B = 2, S = 2048, E = 1024, H = 16;
    const int M = B * S;                 // 4096

    const float* hidden = (const float*)d_in[0];
    const float* wqkv   = (const float*)d_in[1];
    const float* wout   = (const float*)d_in[2];
    float* out = (float*)d_out;

    char* ws = (char*)d_ws;
    size_t off = 0;
    auto alloc = [&](size_t bytes) -> void* {
        void* p = ws + off;
        off += (bytes + 255) & ~(size_t)255;
        return p;
    };
    half_t* hidden_h = (half_t*)alloc((size_t)M * E * 2);        // 8 MB
    half_t* wqkv_h   = (half_t*)alloc((size_t)3 * E * E * 2);    // 6 MB
    half_t* wout_h   = (half_t*)alloc((size_t)E * E * 2);        // 2 MB
    half_t* qkv_h    = (half_t*)alloc((size_t)M * 3 * E * 2);    // 24 MB
    half_t* Qh       = (half_t*)alloc((size_t)M * E * 2);        // 8 MB (B*H*S*64)
    half_t* Kh       = (half_t*)alloc((size_t)M * E * 2);        // 8 MB
    half_t* Vt       = (half_t*)alloc((size_t)M * E * 2);        // 8 MB
    half_t* ao       = (half_t*)alloc((size_t)M * E * 2);        // 8 MB

    f32_to_f16_kernel<<<(M * E / 8 + 255) / 256, 256, 0, stream>>>(hidden, hidden_h, M * E / 8);
    f32_to_f16_kernel<<<(3 * E * E / 8 + 255) / 256, 256, 0, stream>>>(wqkv, wqkv_h, 3 * E * E / 8);
    f32_to_f16_kernel<<<(E * E / 8 + 255) / 256, 256, 0, stream>>>(wout, wout_h, E * E / 8);

    // QKV projection: (4096 x 1024) x (3072 x 1024)^T -> f16
    gemm_wt_kernel<half_t><<<dim3(3 * E / 64, M / 256), 256, 0, stream>>>(
        hidden_h, wqkv_h, qkv_h, M, 3 * E, E);

    // RoPE + split heads + V transpose
    rope_split_kernel<<<(B * H * S * 64) / 256, 256, 0, stream>>>(qkv_h, Qh, Kh, Vt);

    // Flash attention: 4096 q-tiles, 4 waves per block
    attn_kernel<<<(B * H * (S / 16)) / 4, 128, 0, stream>>>(Qh, Kh, Vt, ao);

    // Output projection: (4096 x 1024) x (1024 x 1024)^T -> f32 d_out
    gemm_wt_kernel<float><<<dim3(E / 64, M / 256), 256, 0, stream>>>(
        ao, wout_h, out, M, E, E);
}
